// CircleDotFormer_14757507629328
// MI455X (gfx1250) — matrix operand order repaired and checked by reading, not compiled, not run on hardware
//
#include <hip/hip_runtime.h>
#include <hip/hip_bf16.h>
#include <math.h>

#define NN 50000
#define EE 800000
#define BB 256
#define HH 64

typedef _Float16 half_t;
typedef __attribute__((ext_vector_type(16))) _Float16 v16h;
typedef __attribute__((ext_vector_type(8)))  float    v8f;

// ---------- helpers ----------
static __device__ __forceinline__ unsigned enc_f32(float f) {
  unsigned u = __float_as_uint(f);
  unsigned mask = (u & 0x80000000u) ? 0xFFFFFFFFu : 0x80000000u;
  return u ^ mask;                      // order-preserving float->uint
}
static __device__ __forceinline__ float dec_f32(unsigned e) {
  unsigned u = (e & 0x80000000u) ? (e ^ 0x80000000u) : ~e;
  return __uint_as_float(u);
}
static __device__ __forceinline__ float silu_f(float v) {
  return v / (1.0f + expf(-v));
}

// ---------- generic fill ----------
__global__ void fill_kernel(float* p, float v, long long n) {
  long long i = (long long)blockIdx.x * blockDim.x + threadIdx.x;
  if (i < n) p[i] = v;
}

// ---------- pack f32 weight [K,Nc] row-major -> WMMA B-fragment f16 layout ----------
// layout: packed[((kc*tilesN + tn)*32 + lane)*16 + j]
//   lane = ((k%32)/16)*16 + (n%16), j = k%16, kc = k/32, tn = n/16
__global__ void pack_w_kernel(const float* __restrict__ W, half_t* __restrict__ out,
                              int K, int Nc) {
  int i = blockIdx.x * blockDim.x + threadIdx.x;
  if (i >= K * Nc) return;
  int k = i / Nc, n = i % Nc;
  int kc = k >> 5;
  int hi = (k >> 4) & 1;
  int j  = k & 15;
  int tn = n >> 4;
  int lane = hi * 16 + (n & 15);
  int tilesN = Nc >> 4;
  out[(((size_t)kc * tilesN + tn) * 32 + lane) * 16 + j] = (half_t)W[i];
}

// ---------- WMMA GEMM, N-tiled per wave ----------
// One wave owns a full 16-row M panel: loads each A fragment once, applies it to
// all TN column tiles (TN accumulators).  C = act(A[M, KC*32] @ Bp + bias).
// act: 0 = identity, 1 = tanh.
template <int KC, int TN>
__global__ void gemm_wmma_tile_kernel(const float* __restrict__ A,
                                      const half_t* __restrict__ Bp,
                                      const float* __restrict__ bias,
                                      float* __restrict__ C, int M, int act) {
  constexpr int K  = KC * 32;
  constexpr int Nc = TN * 16;
  int wave = blockIdx.x * (blockDim.x >> 5) + (threadIdx.x >> 5);
  int lane = threadIdx.x & 31;
  int tilesM = M >> 4;
  if (wave >= tilesM) return;            // wave-uniform: EXEC stays all-ones for WMMA

  int mrow = lane & 15;                  // A row within tile; also D column within tile
  int hi   = lane >> 4;
  const float* Arow = A + (size_t)(wave * 16 + mrow) * K;

  v8f acc[TN] = {};
#pragma unroll
  for (int kc = 0; kc < KC; ++kc) {
    v16h a;
    const float* p0 = Arow + kc * 32 + hi * 16;
    const float* p1 = Arow + kc * 32 + hi * 16 + 8;
#pragma unroll
    for (int j = 0; j < 8; ++j) {
      a[j]     = (half_t)p0[j];
      a[8 + j] = (half_t)p1[j];
    }
#pragma unroll
    for (int tn = 0; tn < TN; ++tn) {
      v16h b = *(const v16h*)(Bp + (((size_t)kc * TN + tn) * 32 + lane) * 16);
      acc[tn] = __builtin_amdgcn_wmma_f32_16x16x32_f16(false, a, false, b,
                                                       (short)0, acc[tn], false, false);
    }
  }

#pragma unroll
  for (int tn = 0; tn < TN; ++tn) {
    float bval = bias[tn * 16 + mrow];
#pragma unroll
    for (int r = 0; r < 8; ++r) {
      float v = acc[tn][r] + bval;
      if (act == 1) v = tanhf(v);
      C[(size_t)(wave * 16 + r + 8 * hi) * Nc + tn * 16 + mrow] = v;
    }
  }
}

// ---------- embed: h = x[N,4] @ W[4,64] + b ----------
__global__ void embed_kernel(const float* __restrict__ x, const float* __restrict__ W,
                             const float* __restrict__ b, float* __restrict__ h) {
  int i = blockIdx.x * blockDim.x + threadIdx.x;
  if (i >= NN * HH) return;
  int n = i >> 6, j = i & 63;
  const float* xr = x + n * 4;
  float s = b[j];
#pragma unroll
  for (int q = 0; q < 4; ++q) s += xr[q] * W[q * HH + j];
  h[i] = s;
}

// ---------- edge score + segment max ----------
__global__ void edge_score_kernel(const float* __restrict__ xl, const float* __restrict__ xr,
                                  const float* __restrict__ We, const float* __restrict__ att,
                                  const float* __restrict__ eattr,
                                  const int* __restrict__ src, const int* __restrict__ dst,
                                  float* __restrict__ escore, unsigned* __restrict__ emax) {
  int e = blockIdx.x * blockDim.x + threadIdx.x;
  if (e >= EE) return;
  int s = src[e], d = dst[e];
  float w = eattr[e];
  const float* pl = xl + (size_t)s * HH;
  const float* pr = xr + (size_t)d * HH;
  float sc = 0.0f;
#pragma unroll
  for (int h = 0; h < HH; ++h) {
    float m = pl[h] + pr[h] + w * We[h];
    m = m > 0.0f ? m : 0.2f * m;          // leaky_relu(0.2)
    sc += m * att[h];
  }
  escore[e] = sc;
  atomicMax(emax + d, enc_f32(sc));
}

// ---------- exp(e - emax[dst]) + segment denom ----------
__global__ void edge_exp_kernel(float* __restrict__ escore, const int* __restrict__ dst,
                                const unsigned* __restrict__ emax, float* __restrict__ den) {
  int e = blockIdx.x * blockDim.x + threadIdx.x;
  if (e >= EE) return;
  int d = dst[e];
  float ex = expf(escore[e] - dec_f32(emax[d]));
  escore[e] = ex;
  atomicAdd(den + d, ex);
}

// ---------- scatter-add alpha * xl[src] into aggr[dst] (16 threads per edge) ----------
__global__ void edge_scatter_kernel(const float* __restrict__ ex, const float* __restrict__ den,
                                    const int* __restrict__ src, const int* __restrict__ dst,
                                    const float* __restrict__ xl, float* __restrict__ aggr) {
  long long tid = (long long)blockIdx.x * blockDim.x + threadIdx.x;
  long long e = tid >> 4;
  if (e >= EE) return;
  int q = (int)(tid & 15);
  int s = src[e], d = dst[e];
  float alpha = ex[e] / den[d];
  const float4 xv = *(const float4*)(xl + (size_t)s * HH + q * 4);
  float* op = aggr + (size_t)d * HH + q * 4;
  atomicAdd(op + 0, alpha * xv.x);
  atomicAdd(op + 1, alpha * xv.y);
  atomicAdd(op + 2, alpha * xv.z);
  atomicAdd(op + 3, alpha * xv.w);
}

// ---------- h = silu(aggr + bias) ----------
__global__ void node_finalize_kernel(const float* __restrict__ aggr, const float* __restrict__ bias,
                                     float* __restrict__ h) {
  int i = blockIdx.x * blockDim.x + threadIdx.x;
  if (i >= NN * HH) return;
  h[i] = silu_f(aggr[i] + bias[i & 63]);
}

// ---------- RK4 helpers ----------
__global__ void rk4_step_kernel(float* __restrict__ kacc, float* __restrict__ ystage,
                                const float* __restrict__ h, const float* __restrict__ k,
                                float wk, float cs, int n, int writeStage) {
  int i = blockIdx.x * blockDim.x + threadIdx.x;
  if (i >= n) return;
  float kv = k[i];
  kacc[i] += wk * kv;
  if (writeStage) ystage[i] = h[i] + cs * kv;
}
__global__ void vec_add_kernel(float* __restrict__ a, const float* __restrict__ b, int n) {
  int i = blockIdx.x * blockDim.x + threadIdx.x;
  if (i < n) a[i] += b[i];
}

// ---------- pooling ----------
__global__ void pool_kernel(const float* __restrict__ h, const int* __restrict__ batch,
                            float* __restrict__ gsum, unsigned* __restrict__ gmax,
                            float* __restrict__ gcnt) {
  int i = blockIdx.x * blockDim.x + threadIdx.x;
  if (i >= NN * HH) return;
  int n = i >> 6, j = i & 63;
  int b = batch[n];
  float v = h[i];
  atomicAdd(gsum + (size_t)b * HH + j, v);
  atomicMax(gmax + (size_t)b * HH + j, enc_f32(v));
  if (j == 0) atomicAdd(gcnt + b, 1.0f);
}
__global__ void pool_finalize_kernel(const float* __restrict__ gsum, const unsigned* __restrict__ gmax,
                                     const float* __restrict__ gcnt, float* __restrict__ g) {
  int i = blockIdx.x * blockDim.x + threadIdx.x;
  if (i >= BB * HH) return;
  int b = i >> 6, j = i & 63;
  float cnt = gcnt[b];
  float mean = gsum[i] / fmaxf(cnt, 1.0f);
  float mx = (cnt > 0.0f) ? dec_f32(gmax[i]) : 0.0f;
  g[(size_t)b * (2 * HH) + j] = mean;
  g[(size_t)b * (2 * HH) + HH + j] = mx;
}

// ---------- predictor: out[b] = silu(g @ W1 + b1) @ W2 + b2 ----------
__global__ void predictor_kernel(const float* __restrict__ g, const float* __restrict__ W1,
                                 const float* __restrict__ b1, const float* __restrict__ W2,
                                 const float* __restrict__ b2, float* __restrict__ out) {
  int b = blockIdx.x * blockDim.x + threadIdx.x;
  if (b >= BB) return;
  const float* gb = g + (size_t)b * 128;
  float acc = b2[0];
#pragma unroll 4
  for (int j = 0; j < 32; ++j) {
    float s = b1[j];
    for (int i = 0; i < 128; ++i) s += gb[i] * W1[i * 32 + j];
    acc += silu_f(s) * W2[j];
  }
  out[b] = acc;
}

// ======================================================================
extern "C" void kernel_launch(void* const* d_in, const int* in_sizes, int n_in,
                              void* d_out, int out_size, void* d_ws, size_t ws_size,
                              hipStream_t stream) {
  (void)in_sizes; (void)n_in; (void)out_size; (void)ws_size;
  const float* x      = (const float*)d_in[0];
  const int*   eidx   = (const int*)d_in[1];
  const float* eattr  = (const float*)d_in[2];
  const int*   batch  = (const int*)d_in[3];
  const float* embW   = (const float*)d_in[4];
  const float* embB   = (const float*)d_in[5];
  const float* c1_Wl  = (const float*)d_in[6];
  const float* c1_bl  = (const float*)d_in[7];
  const float* c1_Wr  = (const float*)d_in[8];
  const float* c1_br  = (const float*)d_in[9];
  const float* c1_We  = (const float*)d_in[10];
  const float* c1_att = (const float*)d_in[11];
  const float* c1_b   = (const float*)d_in[12];
  const float* c2_Wl  = (const float*)d_in[13];
  const float* c2_bl  = (const float*)d_in[14];
  const float* c2_Wr  = (const float*)d_in[15];
  const float* c2_br  = (const float*)d_in[16];
  const float* c2_We  = (const float*)d_in[17];
  const float* c2_att = (const float*)d_in[18];
  const float* c2_b   = (const float*)d_in[19];
  const float* odeW1  = (const float*)d_in[20];
  const float* odeB1  = (const float*)d_in[21];
  const float* odeW2  = (const float*)d_in[22];
  const float* odeB2  = (const float*)d_in[23];
  const float* pW1    = (const float*)d_in[24];
  const float* pb1    = (const float*)d_in[25];
  const float* pW2    = (const float*)d_in[26];
  const float* pb2    = (const float*)d_in[27];
  float* out = (float*)d_out;

  const int* src = eidx;        // edge_index[0,:]
  const int* dst = eidx + EE;   // edge_index[1,:]

  // ----- workspace layout (floats; 64-float aligned) -----
  float* wsf = (float*)d_ws;
  size_t off = 0;
  auto alloc = [&](size_t n) { float* p = wsf + off; off += (n + 63) & ~(size_t)63; return p; };
  float*  h    = alloc((size_t)NN * HH);
  float*  t    = alloc((size_t)NN * 2 * HH);   // ODE hidden; aliases xl|xr during convs
  float*  xl   = t;
  float*  xr   = t + (size_t)NN * HH;
  float*  aggr = alloc((size_t)NN * HH);       // conv aggregation; RK4 ystage
  float*  k    = alloc((size_t)NN * HH);
  float*  kacc = alloc((size_t)NN * HH);
  float*  es   = alloc((size_t)EE);
  unsigned* emax = (unsigned*)alloc((size_t)NN);
  float*  den  = alloc((size_t)NN);
  float*  gsum = alloc((size_t)BB * HH);
  unsigned* gmax = (unsigned*)alloc((size_t)BB * HH);
  float*  gcnt = alloc((size_t)BB);
  float*  g    = alloc((size_t)BB * 2 * HH);
  half_t* packs = (half_t*)alloc(32768 / 2);   // 32768 f16 = 16384 float slots
  half_t* p_c1Wl = packs;
  half_t* p_c1Wr = p_c1Wl + 64 * 64;
  half_t* p_c2Wl = p_c1Wr + 64 * 64;
  half_t* p_c2Wr = p_c2Wl + 64 * 64;
  half_t* p_oW1  = p_c2Wr + 64 * 64;           // 64x128
  half_t* p_oW2  = p_oW1 + 64 * 128;           // 128x64

  const int TB = 256;
  auto grid = [](long long n, int b) { return (unsigned)((n + b - 1) / b); };

  // ----- pack weights to WMMA layout (f16) -----
  pack_w_kernel<<<grid(64 * 64, TB), TB, 0, stream>>>(c1_Wl, p_c1Wl, 64, 64);
  pack_w_kernel<<<grid(64 * 64, TB), TB, 0, stream>>>(c1_Wr, p_c1Wr, 64, 64);
  pack_w_kernel<<<grid(64 * 64, TB), TB, 0, stream>>>(c2_Wl, p_c2Wl, 64, 64);
  pack_w_kernel<<<grid(64 * 64, TB), TB, 0, stream>>>(c2_Wr, p_c2Wr, 64, 64);
  pack_w_kernel<<<grid(64 * 128, TB), TB, 0, stream>>>(odeW1, p_oW1, 64, 128);
  pack_w_kernel<<<grid(128 * 64, TB), TB, 0, stream>>>(odeW2, p_oW2, 128, 64);

  // one wave per 16-row M panel; waves = M/16
  const unsigned gemm_blocks = grid((long long)(NN / 16) * 32, TB);

  // ----- embed -----
  embed_kernel<<<grid((long long)NN * HH, TB), TB, 0, stream>>>(x, embW, embB, h);

  // ----- GATv2 conv (x2) -----
  auto conv = [&](const half_t* pWl, const float* bl, const half_t* pWr, const float* br,
                  const float* We, const float* att, const float* cbias) {
    gemm_wmma_tile_kernel<2, 4><<<gemm_blocks, TB, 0, stream>>>(h, pWl, bl, xl, NN, 0);
    gemm_wmma_tile_kernel<2, 4><<<gemm_blocks, TB, 0, stream>>>(h, pWr, br, xr, NN, 0);
    fill_kernel<<<grid(NN, TB), TB, 0, stream>>>((float*)emax, 0.0f, NN);  // enc-min
    fill_kernel<<<grid(NN, TB), TB, 0, stream>>>(den, 0.0f, NN);
    fill_kernel<<<grid((long long)NN * HH, TB), TB, 0, stream>>>(aggr, 0.0f, (long long)NN * HH);
    edge_score_kernel<<<grid(EE, TB), TB, 0, stream>>>(xl, xr, We, att, eattr, src, dst, es, emax);
    edge_exp_kernel<<<grid(EE, TB), TB, 0, stream>>>(es, dst, emax, den);
    edge_scatter_kernel<<<grid((long long)EE * 16, TB), TB, 0, stream>>>(es, den, src, dst, xl, aggr);
    node_finalize_kernel<<<grid((long long)NN * HH, TB), TB, 0, stream>>>(aggr, cbias, h);
  };
  conv(p_c1Wl, c1_bl, p_c1Wr, c1_br, c1_We, c1_att, c1_b);
  conv(p_c2Wl, c2_bl, p_c2Wr, c2_br, c2_We, c2_att, c2_b);

  // ----- RK4: one step t=0 -> 1 -----
  const int NH = NN * HH;
  fill_kernel<<<grid(NH, TB), TB, 0, stream>>>(kacc, 0.0f, NH);
  auto feval = [&](const float* y) {
    gemm_wmma_tile_kernel<2, 8><<<gemm_blocks, TB, 0, stream>>>(y, p_oW1, odeB1, t, NN, 1);  // tanh
    gemm_wmma_tile_kernel<4, 4><<<gemm_blocks, TB, 0, stream>>>(t, p_oW2, odeB2, k, NN, 0);
  };
  feval(h);      // k1
  rk4_step_kernel<<<grid(NH, TB), TB, 0, stream>>>(kacc, aggr, h, k, 1.0f / 6.0f, 0.5f, NH, 1);
  feval(aggr);   // k2
  rk4_step_kernel<<<grid(NH, TB), TB, 0, stream>>>(kacc, aggr, h, k, 1.0f / 3.0f, 0.5f, NH, 1);
  feval(aggr);   // k3
  rk4_step_kernel<<<grid(NH, TB), TB, 0, stream>>>(kacc, aggr, h, k, 1.0f / 3.0f, 1.0f, NH, 1);
  feval(aggr);   // k4
  rk4_step_kernel<<<grid(NH, TB), TB, 0, stream>>>(kacc, aggr, h, k, 1.0f / 6.0f, 0.0f, NH, 0);
  vec_add_kernel<<<grid(NH, TB), TB, 0, stream>>>(h, kacc, NH);

  // ----- pooling -----
  fill_kernel<<<grid(BB * HH, TB), TB, 0, stream>>>(gsum, 0.0f, BB * HH);
  fill_kernel<<<grid(BB * HH, TB), TB, 0, stream>>>((float*)gmax, 0.0f, BB * HH);
  fill_kernel<<<grid(BB, TB), TB, 0, stream>>>(gcnt, 0.0f, BB);
  pool_kernel<<<grid((long long)NN * HH, TB), TB, 0, stream>>>(h, batch, gsum, gmax, gcnt);
  pool_finalize_kernel<<<grid(BB * HH, TB), TB, 0, stream>>>(gsum, gmax, gcnt, g);

  // ----- predictor -----
  predictor_kernel<<<grid(BB, TB), TB, 0, stream>>>(g, pW1, pb1, pW2, pb2, out);
}
